// MacheteLayer_86766929313859
// MI455X (gfx1250) — matrix operand order repaired and checked by reading, not compiled, not run on hardware
//
#include <hip/hip_runtime.h>
#include <hip/hip_bf16.h>

// W4A16 GEMM for gfx1250: out = x(bf16) @ dequant(int4 B, bf16 group scales)
//  - V_WMMA_F32_16X16X32_BF16, wave tile 64M x 64N (16 accumulators, Mf=4 x Nf=4)
//    -> ~140 VALU vs 16 WMMAs per k-step: VALU and XDL pipes balanced.
//  - int4 -> bf16 via interleaved magic-number: ((p>>4i)&0x000F000F)|0x43004300
//    gives bf16 pair (128+nib_i, 128+nib_{i+4}); one v_pk_fma_bf16 applies (w-136)*s.
//  - A K-slots permuted {0,4,1,5,2,6,3,7} per 8-chunk with v_perm_b32 to match.

typedef __attribute__((ext_vector_type(16))) __bf16 v16bf;
typedef __attribute__((ext_vector_type(8)))  float  v8f;
typedef __attribute__((ext_vector_type(2)))  __bf16 v2bf;

#define MM 256
#define KK 4096
#define NN 12288
#define GG 128   // quant group size along K

__device__ __forceinline__ unsigned short f32_to_bf16_rne(float f) {
    unsigned u = __builtin_bit_cast(unsigned, f);
    u += 0x7fffu + ((u >> 16) & 1u);
    return (unsigned short)(u >> 16);
}

__global__ __launch_bounds__(128) void machete_wmma_bf16(
    const __bf16* __restrict__ X,     // [M, K] row-major
    const int*    __restrict__ Bq,    // [K/8, N] row-major, 8 nibbles along K per int32
    const __bf16* __restrict__ S,     // [K/128, N] row-major
    unsigned short* __restrict__ Out) // [M, N] bf16 bits
{
    const int lane   = threadIdx.x & 31;
    const int wave   = threadIdx.x >> 5;    // 4 waves, tiled along N
    const int laneM  = lane & 15;
    const int laneHi = lane >> 4;           // lane half

    const int m0 = blockIdx.y * 64;                // wave covers rows m0..m0+63
    const int n0 = blockIdx.x * 256 + wave * 64;   // wave covers cols n0..n0+63

    // A: lanes 0-15 hold K chunks [0..7],[16..23]; lanes 16-31 hold [8..15],[24..31].
    const __bf16* xrow = X + (size_t)(m0 + laneM) * KK + laneHi * 8;

    // B: lane column n; lanes 0-15 carry klocal 0..7,16..23 (rows k/8, k/8+2),
    //    lanes 16-31 carry klocal 8..15,24..31 (rows k/8+1, k/8+3).
    int ncol[4];
#pragma unroll
    for (int f = 0; f < 4; ++f) ncol[f] = n0 + 16 * f + laneM;

    const int* bcol[4];
#pragma unroll
    for (int f = 0; f < 4; ++f) bcol[f] = Bq + ncol[f];

    v8f acc[4][4] = {};   // 16 x (16x16 f32): 64M x 64N

    for (int g = 0; g < KK / GG; ++g) {
        // Per-group dequant constants: (w * s) + (-136*s) == (nib-8)*s.
        v2bf spk[4], cpk[4];
#pragma unroll
        for (int f = 0; f < 4; ++f) {
            __bf16 sv = S[(size_t)g * NN + ncol[f]];
            __bf16 cv = (__bf16)(-136.0f * (float)sv);
            spk[f] = (v2bf){sv, sv};
            cpk[f] = (v2bf){cv, cv};
        }

#pragma unroll
        for (int kk = 0; kk < 4; ++kk) {
            const int k = g * GG + kk * 32;

            // ---- 4 A fragments: 2x 16B loads each, interleave K {0,4,1,5,2,6,3,7} ----
            v16bf Aw[4];
#pragma unroll
            for (int a = 0; a < 4; ++a) {
                const __bf16* xr = xrow + (size_t)(a * 16) * KK;
                uint4 q0 = *(const uint4*)(xr + k);        // k+koff+0..7
                uint4 q1 = *(const uint4*)(xr + k + 16);   // k+koff+16..23
                union { v16bf v; unsigned u[8]; } A;
                A.u[0] = __builtin_amdgcn_perm(q0.z, q0.x, 0x05040100u); // (k0,k4)
                A.u[1] = __builtin_amdgcn_perm(q0.z, q0.x, 0x07060302u); // (k1,k5)
                A.u[2] = __builtin_amdgcn_perm(q0.w, q0.y, 0x05040100u); // (k2,k6)
                A.u[3] = __builtin_amdgcn_perm(q0.w, q0.y, 0x07060302u); // (k3,k7)
                A.u[4] = __builtin_amdgcn_perm(q1.z, q1.x, 0x05040100u);
                A.u[5] = __builtin_amdgcn_perm(q1.z, q1.x, 0x07060302u);
                A.u[6] = __builtin_amdgcn_perm(q1.w, q1.y, 0x05040100u);
                A.u[7] = __builtin_amdgcn_perm(q1.w, q1.y, 0x07060302u);
                Aw[a] = A.v;
            }

            const int    r0  = (k >> 3) + laneHi;
            const size_t ro0 = (size_t)r0 * NN;
            const size_t ro1 = ro0 + 2 * NN;

            // ---- 4 B fragments: dequant once, feed 4 WMMAs each ----
#pragma unroll
            for (int f = 0; f < 4; ++f) {
                const unsigned p0 = (unsigned)bcol[f][ro0];  // klocal 0..7 (this lane half)
                const unsigned p1 = (unsigned)bcol[f][ro1];  // klocal 16..23 (this lane half)

                union { v16bf v; v2bf h[8]; } Bf;
#pragma unroll
                for (int i = 0; i < 4; ++i) {
                    // bf16 pair (128+nib_i, 128+nib_{i+4}) in one and-or
                    unsigned w0 = ((p0 >> (4 * i)) & 0x000F000Fu) | 0x43004300u;
                    unsigned w1 = ((p1 >> (4 * i)) & 0x000F000Fu) | 0x43004300u;
                    Bf.h[i]     = __builtin_bit_cast(v2bf, w0) * spk[f] + cpk[f];
                    Bf.h[4 + i] = __builtin_bit_cast(v2bf, w1) * spk[f] + cpk[f];
                }

#pragma unroll
                for (int a = 0; a < 4; ++a) {
                    acc[a][f] = __builtin_amdgcn_wmma_f32_16x16x32_bf16(
                        false, Aw[a], false, Bf.v, (short)0, acc[a][f], false, false);
                }
            }
        }
    }

    // ---- writeout: VGPR r holds row (r + 8*laneHi), col = laneM + 16*f ----
#pragma unroll
    for (int a = 0; a < 4; ++a) {
#pragma unroll
        for (int f = 0; f < 4; ++f) {
            const int nc = ncol[f];
#pragma unroll
            for (int r = 0; r < 8; ++r) {
                const int mrow = m0 + a * 16 + r + laneHi * 8;
                Out[(size_t)mrow * NN + nc] = f32_to_bf16_rne(acc[a][f][r]);
            }
        }
    }
}

extern "C" void kernel_launch(void* const* d_in, const int* in_sizes, int n_in,
                              void* d_out, int out_size, void* d_ws, size_t ws_size,
                              hipStream_t stream) {
    const __bf16* X  = (const __bf16*)d_in[0];
    const int*    Bq = (const int*)d_in[1];
    const __bf16* S  = (const __bf16*)d_in[2];
    unsigned short* Out = (unsigned short*)d_out;

    dim3 grid(NN / 256, MM / 64, 1);   // 48 x 4 = 192 workgroups
    dim3 block(128, 1, 1);             // 4 wave32s along N
    hipLaunchKernelGGL(machete_wmma_bf16, grid, block, 0, stream, X, Bq, S, Out);
}
